// Attention_60266981097535
// MI455X (gfx1250) — matrix operand order repaired and checked by reading, not compiled, not run on hardware
//
#include <hip/hip_runtime.h>
#include <hip/hip_bf16.h>
#include <math.h>

// Problem constants (reference: B,T,DIM,H = 4,1024,256,8 ; D = 32)
constexpr int NB = 4, NT = 1024, NDIM = 256, NH = 8, HD = 32;

typedef _Float16 half_t;
typedef __attribute__((ext_vector_type(16))) _Float16 v16h;
typedef __attribute__((ext_vector_type(8)))  _Float16 v8h;
typedef __attribute__((ext_vector_type(8)))  float    v8f;
typedef __attribute__((ext_vector_type(4)))  int      v4i;

__device__ __forceinline__ float gelu_exact(float x) {
  return 0.5f * x * (1.0f + erff(x * 0.70710678118654752f));
}

__device__ __forceinline__ v8f wmma16(v16h a, v16h b, v8f c) {
  // D = A(16x32 f16) x B(32x16 f16) + C(16x16 f32)
  return __builtin_amdgcn_wmma_f32_16x16x32_f16(false, a, false, b, (short)0, c,
                                                false, false);
}

// A-matrix (16x32 f16) loader, row-major source. ISA layout: lane l (m=l&15,
// g=l>>4) holds row m, K = {8g..8g+7} in regs 0-3 and {16+8g..16+8g+7} in 4-7.
__device__ __forceinline__ v16h load_A16(const half_t* base, int stride) {
  int l = threadIdx.x & 31, m = l & 15, g = l >> 4;
  const half_t* row = base + (size_t)m * stride;
  v8h lo = *(const v8h*)(row + 8 * g);
  v8h hi = *(const v8h*)(row + 16 + 8 * g);
  v16h r;
#pragma unroll
  for (int i = 0; i < 8; ++i) { r[i] = lo[i]; r[8 + i] = hi[i]; }
  return r;
}

// B-matrix (32x16 f16) loader from TRANSPOSED storage baseT[n][K] (strideT
// elems per n-row). Lane l (n=l&15, g=l>>4) holds column n, K = {8g..8g+7} in
// regs 0-3 and {16+8g..16+8g+7} in 4-7.
__device__ __forceinline__ v16h load_B16(const half_t* baseT, int strideT) {
  int l = threadIdx.x & 31, n = l & 15, g = l >> 4;
  const half_t* p = baseT + (size_t)n * strideT;
  v8h lo = *(const v8h*)(p + 8 * g);
  v8h hi = *(const v8h*)(p + 16 + 8 * g);
  v16h r;
#pragma unroll
  for (int i = 0; i < 8; ++i) { r[i] = lo[i]; r[8 + i] = hi[i]; }
  return r;
}

// A-matrix tile from the int32 adjacency (values 0/1, exact in f16).
__device__ __forceinline__ v16h load_A_adj(const int* base) {
  int l = threadIdx.x & 31, m = l & 15, g = l >> 4;
  const int* row = base + (size_t)m * NT;
  const int* p0 = row + 8 * g;
  const int* p1 = row + 16 + 8 * g;
  v4i a0 = *(const v4i*)p0, a1 = *(const v4i*)(p0 + 4);
  v4i b0 = *(const v4i*)p1, b1 = *(const v4i*)(p1 + 4);
  v16h r;
#pragma unroll
  for (int i = 0; i < 4; ++i) {
    r[i]      = (_Float16)(float)a0[i];
    r[4 + i]  = (_Float16)(float)a1[i];
    r[8 + i]  = (_Float16)(float)b0[i];
    r[12 + i] = (_Float16)(float)b1[i];
  }
  return r;
}

// ---------------------------------------------------------------- K0: weights
// Transpose + f16-convert the 256x256 projection weights (Wt[n][k] = W[k][n]).
__global__ __launch_bounds__(256) void k_wconv(const float* Wq, const float* Wk,
                                               const float* Wv, half_t* Wqt,
                                               half_t* Wkt, half_t* Wvt) {
  int i = blockIdx.x * 256 + threadIdx.x;   // i = k*256 + n
  int k = i >> 8, n = i & 255;
  Wqt[(size_t)n * NDIM + k] = (half_t)Wq[i];
  Wkt[(size_t)n * NDIM + k] = (half_t)Wk[i];
  Wvt[(size_t)n * NDIM + k] = (half_t)Wv[i];
}

// -------------------------------------------------------- K1: X @ W_g per head
// XQt/XKt/XVt[b,h,d,t] = sum_e x[b,t,h*32+e] * W[e,d], stored f16 TRANSPOSED
// (d-major) so GEMM B-tile loads are contiguous.
__global__ __launch_bounds__(256) void k_xw(const float* x, const float* Wqg,
                                            const float* Wkg, const float* Wvg,
                                            half_t* XQt, half_t* XKt,
                                            half_t* XVt) {
  __shared__ float xs[32][32];
  __shared__ float wq[32][32], wk[32][32], wv[32][32];
  int tid = threadIdx.x;
  int blk = blockIdx.x;            // NB*NH*(NT/32) blocks
  int tchunk = blk & 31;
  int bh = blk >> 5;
  int b = bh >> 3, h = bh & 7;
  int t0 = tchunk * 32;
#pragma unroll
  for (int k = 0; k < 4; ++k) {
    int idx = tid + k * 256;
    int r = idx >> 5, c = idx & 31;
    wq[r][c] = Wqg[idx];
    wk[r][c] = Wkg[idx];
    wv[r][c] = Wvg[idx];
    xs[r][c] = x[((size_t)(b * NT + t0 + r)) * NDIM + h * HD + c];
  }
  __syncthreads();
  int ts = tid & 31, dq = tid >> 5;
#pragma unroll
  for (int j = 0; j < 4; ++j) {
    int d = dq + 8 * j;
    float aq = 0.f, ak = 0.f, av = 0.f;
#pragma unroll
    for (int e = 0; e < 32; ++e) {
      float xv = xs[ts][e];
      aq += xv * wq[e][d];
      ak += xv * wk[e][d];
      av += xv * wv[e][d];
    }
    size_t o = ((size_t)(b * NH + h) * HD + d) * NT + t0 + ts;
    XQt[o] = (half_t)aq;
    XKt[o] = (half_t)ak;
    XVt[o] = (half_t)av;
  }
}

// --------------------------------------------- K2: GCN  relu(adj @ (x W)) WMMA
// Per (b,h): M=1024, N=32, K=1024. A = adj tile (shared by q and k paths).
__global__ __launch_bounds__(256) void k_gcn(const int* adj, const half_t* XQt,
                                             const half_t* XKt, half_t* qg,
                                             half_t* kg) {
  int wave = blockIdx.x * 8 + (threadIdx.x >> 5);  // 0..2047
  int mtile = wave & 63;
  int bh = wave >> 6;
  int b = bh >> 3, h = bh & 7;
  int s0 = mtile * 16;
  const int* adjb = adj + ((size_t)(b * NH + h) * NT + s0) * NT;
  const half_t* xq = XQt + (size_t)(b * NH + h) * HD * NT;
  const half_t* xk = XKt + (size_t)(b * NH + h) * HD * NT;
  v8f cq0 = {}, cq1 = {}, ck0 = {}, ck1 = {};
  for (int k0 = 0; k0 < NT; k0 += 32) {
    v16h A = load_A_adj(adjb + k0);
    v16h bq0 = load_B16(xq + k0, NT);
    v16h bq1 = load_B16(xq + 16 * NT + k0, NT);
    v16h bk0 = load_B16(xk + k0, NT);
    v16h bk1 = load_B16(xk + 16 * NT + k0, NT);
    cq0 = wmma16(A, bq0, cq0);
    cq1 = wmma16(A, bq1, cq1);
    ck0 = wmma16(A, bk0, ck0);
    ck1 = wmma16(A, bk1, ck1);
  }
  int l = threadIdx.x & 31, n = l & 15, g = l >> 4;
  half_t* qo = qg + ((size_t)(b * NH + h) * NT + s0) * HD;
  half_t* ko = kg + ((size_t)(b * NH + h) * NT + s0) * HD;
#pragma unroll
  for (int r = 0; r < 8; ++r) {
    int row = r + 8 * g;
    qo[row * HD + n]      = (half_t)fmaxf(cq0[r], 0.f);
    qo[row * HD + 16 + n] = (half_t)fmaxf(cq1[r], 0.f);
    ko[row * HD + n]      = (half_t)fmaxf(ck0[r], 0.f);
    ko[row * HD + 16 + n] = (half_t)fmaxf(ck1[r], 0.f);
  }
}

// ---------------------------------------- K3: proj_q / proj_k  (4096x256x256)
// A rows = (b,t) with K = h*32+d read from head-major qg/kg; B = transposed W.
__global__ __launch_bounds__(256) void k_proj(const half_t* qg, const half_t* kg,
                                              const half_t* Wqt,
                                              const half_t* Wkt, half_t* q2,
                                              half_t* k2) {
  int wave = blockIdx.x * 8 + (threadIdx.x >> 5);  // 0..4095
  int ntile = wave & 7;
  int mtile = (wave >> 3) & 255;
  int path = wave >> 11;
  const half_t* src = path ? kg : qg;
  const half_t* Wt = path ? Wkt : Wqt;
  half_t* dst = path ? k2 : q2;
  int M0 = mtile * 16, n0 = ntile * 32;
  int l = threadIdx.x & 31, m = l & 15, g = l >> 4, n = l & 15;
  int Mrow = M0 + m, b = Mrow >> 10, t = Mrow & 1023;
  v8f c0 = {}, c1 = {};
  for (int k0 = 0; k0 < NDIM; k0 += 32) {
    int h = k0 >> 5;
    const half_t* row = src + ((size_t)(b * NH + h) * NT + t) * HD;
    v8h lo = *(const v8h*)(row + 8 * g);
    v8h hi = *(const v8h*)(row + 16 + 8 * g);
    v16h A;
#pragma unroll
    for (int i = 0; i < 8; ++i) { A[i] = lo[i]; A[8 + i] = hi[i]; }
    v16h B0 = load_B16(Wt + (size_t)n0 * NDIM + k0, NDIM);
    v16h B1 = load_B16(Wt + (size_t)(n0 + 16) * NDIM + k0, NDIM);
    c0 = wmma16(A, B0, c0);
    c1 = wmma16(A, B1, c1);
  }
#pragma unroll
  for (int r = 0; r < 8; ++r) {
    int row = M0 + r + 8 * g;
    dst[(size_t)row * NDIM + n0 + n]      = (half_t)c0[r];
    dst[(size_t)row * NDIM + n0 + 16 + n] = (half_t)c1[r];
  }
}

// ------------------------------------- K4: R0 = gelu(concat(q,k) @ Wkf) [b,t,h]
__global__ __launch_bounds__(256) void k_r0(const half_t* q2, const half_t* k2,
                                            const float* Wkf, float* R0b) {
  __shared__ float wl[512 * NH];
  int tid = threadIdx.x;
  for (int i = tid; i < 512 * NH; i += 256) wl[i] = Wkf[i];
  __syncthreads();
  int bt = blockIdx.x * 256 + tid;
  const half_t* qr = q2 + (size_t)bt * NDIM;
  const half_t* kr = k2 + (size_t)bt * NDIM;
  float acc[NH] = {};
  for (int j = 0; j < NDIM; ++j) {
    float qv = (float)qr[j];
    const float* w = &wl[j * NH];
#pragma unroll
    for (int h = 0; h < NH; ++h) acc[h] += qv * w[h];
  }
  for (int j = 0; j < NDIM; ++j) {
    float kv = (float)kr[j];
    const float* w = &wl[(NDIM + j) * NH];
#pragma unroll
    for (int h = 0; h < NH; ++h) acc[h] += kv * w[h];
  }
#pragma unroll
  for (int h = 0; h < NH; ++h) R0b[(size_t)bt * NH + h] = gelu_exact(acc[h]);
}

// ------------ K5: R = Gram(R0); Rf[b,h,t] = sigmoid((R @ Wkf2)/sparse_D) -----
__global__ __launch_bounds__(256) void k_rf(const float* R0b, const float* Wkf2,
                                            const float* sD, float* Rfb) {
  __shared__ float Racc[NH * NH];
  int tid = threadIdx.x, b = blockIdx.x;
  if (tid < NH * NH) Racc[tid] = 0.f;
  __syncthreads();
  for (int t = tid; t < NT; t += 256) {
    float r[NH];
#pragma unroll
    for (int h = 0; h < NH; ++h) r[h] = R0b[((size_t)(b * NT + t)) * NH + h];
#pragma unroll
    for (int h = 0; h < NH; ++h)
#pragma unroll
      for (int k = 0; k < NH; ++k) atomicAdd(&Racc[h * NH + k], r[h] * r[k]);
  }
  __syncthreads();
  for (int t = tid; t < NT; t += 256) {
    for (int h = 0; h < NH; ++h) {
      float s = 0.f;
#pragma unroll
      for (int k = 0; k < NH; ++k) s += Racc[h * NH + k] * Wkf2[k * NT + t];
      s /= sD[h * NT + t];
      Rfb[(size_t)(b * NH + h) * NT + t] = 1.f / (1.f + expf(-s));
    }
  }
}

// -------------------- K6: fused scores->mix->mask->softmax->(p @ xv)  (flash)
// Block = (b, 16-row s-tile), 8 waves. Wave w produces scores for head h=w
// into LDS; consumes all 8 heads for output head l=w. Online softmax; p-tile
// staged through per-wave LDS to convert C-layout -> A-layout for the pv WMMA.
__global__ __launch_bounds__(256) void k_attn(const half_t* q2, const half_t* k2,
                                              const half_t* XVt, const int* adj,
                                              const float* Rf, const float* RM,
                                              half_t* vg) {
  __shared__ float sc[NH][16][32];    // scaled per-head scores
  __shared__ half_t pst[NH][16][32];  // per-wave p staging (D->A layout)
  int tid = threadIdx.x;
  int w = tid >> 5;
  int l = tid & 31, n = l & 15, g = l >> 4;
  int b = blockIdx.x >> 6;
  int s0 = (blockIdx.x & 63) << 4;

  v16h QA = load_A16(q2 + ((size_t)(b * NT + s0)) * NDIM + w * HD, NDIM);

  float rm[NH];
#pragma unroll
  for (int h = 0; h < NH; ++h) rm[h] = RM[w * NH + h];
  float rs[8];
#pragma unroll
  for (int r = 0; r < 8; ++r) rs[r] = Rf[(size_t)(b * NH + w) * NT + s0 + r + 8 * g];

  v8f acc0 = {}, acc1 = {};
  float mrun[8], srun[8];
#pragma unroll
  for (int r = 0; r < 8; ++r) { mrun[r] = -3.0e38f; srun[r] = 0.f; }

  const half_t* kbase = k2 + ((size_t)(b * NT)) * NDIM + w * HD;
  const half_t* vbase = XVt + (size_t)(b * NH + w) * HD * NT;
  const int* adjb = adj + ((size_t)(b * NH + w) * NT + s0) * NT;
  const float* rtb = Rf + (size_t)(b * NH + w) * NT;
  const float scale = 0.0625f;  // 256^-0.5

  for (int t0 = 0; t0 < NT; t0 += 32) {
    // ---- scores for head w (two 16-col tiles), leaky_relu + R scaling
    v16h KB0 = load_B16(kbase + (size_t)t0 * NDIM, NDIM);
    v16h KB1 = load_B16(kbase + (size_t)(t0 + 16) * NDIM, NDIM);
    v8f z = {};
    v8f S0 = wmma16(QA, KB0, z);
    v8f S1 = wmma16(QA, KB1, z);
    float rt0 = rtb[t0 + n];
    float rt1 = rtb[t0 + 16 + n];
#pragma unroll
    for (int r = 0; r < 8; ++r) {
      float x0 = S0[r] * scale; x0 = x0 > 0.f ? x0 : 0.01f * x0;
      float x1 = S1[r] * scale; x1 = x1 > 0.f ? x1 : 0.01f * x1;
      sc[w][r + 8 * g][n]      = x0 * rs[r] * rt0;
      sc[w][r + 8 * g][16 + n] = x1 * rs[r] * rt1;
    }
    __syncthreads();
    // ---- mix heads with randomatrix, mask with adj (output head l=w)
    float x0[8], x1[8];
#pragma unroll
    for (int r = 0; r < 8; ++r) {
      int row = r + 8 * g;
      float a0 = 0.f, a1 = 0.f;
#pragma unroll
      for (int h = 0; h < NH; ++h) {
        a0 += rm[h] * sc[h][row][n];
        a1 += rm[h] * sc[h][row][16 + n];
      }
      const int* arow = adjb + (size_t)row * NT + t0;
      x0[r] = (arow[n] > 0) ? a0 : -1.0e12f;
      x1[r] = (arow[16 + n] > 0) ? a1 : -1.0e12f;
    }
    __syncthreads();  // sc fully consumed; safe to overwrite next iteration
    // ---- online softmax update + stage p as f16 A-tile
#pragma unroll
    for (int r = 0; r < 8; ++r) {
      float tmax = fmaxf(x0[r], x1[r]);
#pragma unroll
      for (int off = 1; off < 16; off <<= 1)
        tmax = fmaxf(tmax, __shfl_xor(tmax, off, 32));
      float mnew = fmaxf(mrun[r], tmax);
      float corr = expf(mrun[r] - mnew);
      float p0 = expf(x0[r] - mnew);
      float p1 = expf(x1[r] - mnew);
      float rsum = p0 + p1;
#pragma unroll
      for (int off = 1; off < 16; off <<= 1)
        rsum += __shfl_xor(rsum, off, 32);
      srun[r] = srun[r] * corr + rsum;
      mrun[r] = mnew;
      acc0[r] *= corr;
      acc1[r] *= corr;
      pst[w][r + 8 * g][n]      = (half_t)p0;
      pst[w][r + 8 * g][16 + n] = (half_t)p1;
    }
    // ---- p @ xv : re-load p in A layout (wave-local LDS round trip)
    v16h PA = load_A16((const half_t*)&pst[w][0][0], 32);
    v16h VB0 = load_B16(vbase + t0, NT);
    v16h VB1 = load_B16(vbase + 16 * NT + t0, NT);
    acc0 = wmma16(PA, VB0, acc0);
    acc1 = wmma16(PA, VB1, acc1);
  }
  // ---- v = relu(attn @ xv) = relu(acc / srun)
  half_t* vo = vg + ((size_t)(b * NH + w) * NT + s0) * HD;
#pragma unroll
  for (int r = 0; r < 8; ++r) {
    int row = r + 8 * g;
    float inv = 1.f / srun[r];
    vo[row * HD + n]      = (half_t)fmaxf(acc0[r] * inv, 0.f);
    vo[row * HD + 16 + n] = (half_t)fmaxf(acc1[r] * inv, 0.f);
  }
}

// -------------------------- K7: out = gelu(v @ Wv)  (4096x256x256, f32 output)
__global__ __launch_bounds__(256) void k_out(const half_t* vg, const half_t* Wvt,
                                             float* out) {
  int wave = blockIdx.x * 8 + (threadIdx.x >> 5);  // 0..2047
  int ntile = wave & 7;
  int mtile = wave >> 3;
  int M0 = mtile * 16, n0 = ntile * 32;
  int l = threadIdx.x & 31, m = l & 15, g = l >> 4, n = l & 15;
  int Mrow = M0 + m, b = Mrow >> 10, t = Mrow & 1023;
  v8f c0 = {}, c1 = {};
  for (int k0 = 0; k0 < NDIM; k0 += 32) {
    int h = k0 >> 5;
    const half_t* row = vg + ((size_t)(b * NH + h) * NT + t) * HD;
    v8h lo = *(const v8h*)(row + 8 * g);
    v8h hi = *(const v8h*)(row + 16 + 8 * g);
    v16h A;
#pragma unroll
    for (int i = 0; i < 8; ++i) { A[i] = lo[i]; A[8 + i] = hi[i]; }
    v16h B0 = load_B16(Wvt + (size_t)n0 * NDIM + k0, NDIM);
    v16h B1 = load_B16(Wvt + (size_t)(n0 + 16) * NDIM + k0, NDIM);
    c0 = wmma16(A, B0, c0);
    c1 = wmma16(A, B1, c1);
  }
#pragma unroll
  for (int r = 0; r < 8; ++r) {
    int row = M0 + r + 8 * g;
    out[(size_t)row * NDIM + n0 + n]      = gelu_exact(c0[r]);
    out[(size_t)row * NDIM + n0 + 16 + n] = gelu_exact(c1[r]);
  }
}

extern "C" void kernel_launch(void* const* d_in, const int* in_sizes, int n_in,
                              void* d_out, int out_size, void* d_ws,
                              size_t ws_size, hipStream_t stream) {
  const float* x    = (const float*)d_in[0];
  const int*   adj  = (const int*)d_in[1];
  const float* Wqg  = (const float*)d_in[2];
  const float* Wkg  = (const float*)d_in[3];
  const float* Wvg  = (const float*)d_in[4];
  const float* Wq   = (const float*)d_in[5];
  const float* Wk   = (const float*)d_in[6];
  const float* Wv   = (const float*)d_in[7];
  const float* Wkf  = (const float*)d_in[8];
  const float* Wkf2 = (const float*)d_in[9];
  const float* sD   = (const float*)d_in[10];
  const float* RM   = (const float*)d_in[11];
  float* out = (float*)d_out;

  char* ws = (char*)d_ws;
  size_t off = 0;
  auto alloc = [&](size_t bytes) -> void* {
    void* p = ws + off;
    off += (bytes + 255) & ~(size_t)255;
    return p;
  };
  const size_t nBHTD = (size_t)NB * NH * NT * HD;          // 1M elems
  half_t* XQt = (half_t*)alloc(nBHTD * 2);                 // [B,H,D,T]
  half_t* XKt = (half_t*)alloc(nBHTD * 2);
  half_t* XVt = (half_t*)alloc(nBHTD * 2);
  half_t* qg  = (half_t*)alloc(nBHTD * 2);                 // [B,H,T,D]
  half_t* kg  = (half_t*)alloc(nBHTD * 2);
  half_t* q2  = (half_t*)alloc((size_t)NB * NT * NDIM * 2);// [B,T,DIM]
  half_t* k2  = (half_t*)alloc((size_t)NB * NT * NDIM * 2);
  half_t* vg  = (half_t*)alloc(nBHTD * 2);                 // [B,H,T,D]
  half_t* Wqt = (half_t*)alloc((size_t)NDIM * NDIM * 2);
  half_t* Wkt = (half_t*)alloc((size_t)NDIM * NDIM * 2);
  half_t* Wvt = (half_t*)alloc((size_t)NDIM * NDIM * 2);
  float*  R0b = (float*)alloc((size_t)NB * NT * NH * 4);   // [B,T,H]
  float*  Rfb = (float*)alloc((size_t)NB * NH * NT * 4);   // [B,H,T]
  if (off > ws_size) return;  // workspace too small; nothing sane to do

  k_wconv<<<NDIM * NDIM / 256, 256, 0, stream>>>(Wq, Wk, Wv, Wqt, Wkt, Wvt);
  k_xw<<<NB * NH * (NT / 32), 256, 0, stream>>>(x, Wqg, Wkg, Wvg, XQt, XKt, XVt);
  k_gcn<<<NB * NH * (NT / 16) / 8, 256, 0, stream>>>(adj, XQt, XKt, qg, kg);
  k_proj<<<2 * (NB * NT / 16) * (NDIM / 32) / 8, 256, 0, stream>>>(qg, kg, Wqt,
                                                                   Wkt, q2, k2);
  k_r0<<<NB * NT / 256, 256, 0, stream>>>(q2, k2, Wkf, R0b);
  k_rf<<<NB, 256, 0, stream>>>(R0b, Wkf2, sD, Rfb);
  k_attn<<<NB * (NT / 16), 256, 0, stream>>>(q2, k2, XVt, adj, Rfb, RM, vg);
  k_out<<<(NB * NT / 16) * (NDIM / 32) / 8, 256, 0, stream>>>(vg, Wvt, out);
}
